// FusionAwareInterp_70119636075169
// MI455X (gfx1250) — matrix operand order repaired and checked by reading, not compiled, not run on hardware
//
#include <hip/hip_runtime.h>
#include <hip/hip_bf16.h>
#include <math.h>

typedef __attribute__((ext_vector_type(16))) _Float16 v16h;
typedef __attribute__((ext_vector_type(8)))  float    v8f;

#define BSZ   2
#define NPTS  4096
#define CCH   64
#define IMH   128
#define IMW   256
#define QTOT  (IMH * IMW)   // 32768 queries per batch
#define KNN   3

// ---------------------------------------------------------------------------
// Kernel 1: brute-force 3-NN. uv staged interleaved in LDS (32KB, one
// ds_load_b64 broadcast per candidate), one query per thread. The top-3
// insert is guarded by a wave-coherent branch: expected taken fraction over
// 4096 candidates is ~11%, so the common path is dist(3 VALU) + cmp + branch.
// Results are bit-identical to the unguarded cndmask chain.
// ---------------------------------------------------------------------------
__global__ __launch_bounds__(256) void fai_knn_kernel(const float* __restrict__ uv,
                                                      int* __restrict__ knn_out) {
    __shared__ float2 suv[NPTS];

    const int blocksPerBatch = QTOT / 256;                 // 128
    const int b     = blockIdx.x / blocksPerBatch;
    const int qbase = (blockIdx.x % blocksPerBatch) * 256;

    const float* uvb = uv + (size_t)b * 2 * NPTS;
    for (int i = threadIdx.x; i < NPTS; i += 256) {
        suv[i] = make_float2(uvb[i], uvb[NPTS + i]);       // planar -> interleaved
    }
    __syncthreads();

    const int   q  = qbase + threadIdx.x;
    const float fx = (float)(q % IMW);
    const float fy = (float)(q / IMW);

    float d0 = 3.0e38f, d1 = 3.0e38f, d2 = 3.0e38f;
    int   i0 = 0, i1 = 0, i2 = 0;

#pragma unroll 8
    for (int i = 0; i < NPTS; ++i) {
        const float2 p  = suv[i];
        const float  dx = p.x - fx;
        const float  dy = p.y - fy;
        const float  d  = __fmaf_rn(dx, dx, dy * dy);
        if (d < d2) {                       // wave-coherent early-out (~11% taken)
            const bool lt0 = d < d0;
            const bool lt1 = d < d1;
            d2 = lt1 ? d1 : d;
            i2 = lt1 ? i1 : i;
            d1 = lt0 ? d0 : (lt1 ? d : d1);
            i1 = lt0 ? i0 : (lt1 ? i : i1);
            d0 = lt0 ? d  : d0;
            i0 = lt0 ? i  : i0;
        }
    }

    int* o = knn_out + ((size_t)b * QTOT + q) * KNN;
    o[0] = i0; o[1] = i1; o[2] = i2;
}

// ---------------------------------------------------------------------------
// Kernel 2: fused gather + score-net + weighted sum + out_conv (WMMA).
// 256 threads = 8 waves; each wave owns 16 queries (one 16-col WMMA panel).
// Lane l: half = l>>4, qn = l&15 -> query = base + qn.
//   B chunk0 (K-chans 0..31):  lane half=0 supplies chans 0..15 of query qn,
//                              lane half=1 supplies chans 16..31.
//   B chunk1 (K-chans 32..63): half=0 -> 32..47, half=1 -> 48..63.
// final[] is computed straight into B-operand registers (no LDS round-trip).
// A (w_out, f16) is staged in LDS and gathered in the same contiguous K order as B
// (lanes 16-31 hold K=16..31 within each 32-wide chunk).
// ---------------------------------------------------------------------------
__global__ __launch_bounds__(256) void fai_interp_kernel(
        const float* __restrict__ uv,   const float* __restrict__ feat,
        const int*   __restrict__ knn,
        const float* __restrict__ w1,   const float* __restrict__ b1,
        const float* __restrict__ w2,   const float* __restrict__ b2,
        const float* __restrict__ w_out,const float* __restrict__ b_out,
        float* __restrict__ out) {
    __shared__ _Float16 s_wout[CCH * CCH];   // 8 KB, row-major [out_ch][in_ch]
    __shared__ float    s_w2[CCH * 3];
    __shared__ float    s_b2[CCH];
    __shared__ float    s_bout[CCH];
    __shared__ float    s_w1[9];
    __shared__ float    s_b1[3];

    for (int i = threadIdx.x; i < CCH * CCH; i += 256) s_wout[i] = (_Float16)w_out[i];
    for (int i = threadIdx.x; i < CCH * 3;   i += 256) s_w2[i]   = w2[i];
    if (threadIdx.x < CCH) {
        s_b2[threadIdx.x]   = b2[threadIdx.x];
        s_bout[threadIdx.x] = b_out[threadIdx.x];
    }
    if (threadIdx.x < 9) s_w1[threadIdx.x] = w1[threadIdx.x];
    if (threadIdx.x < 3) s_b1[threadIdx.x] = b1[threadIdx.x];
    __syncthreads();

    const int wave = threadIdx.x >> 5;
    const int lane = threadIdx.x & 31;
    const int half = lane >> 4;      // 0 or 1
    const int qn   = lane & 15;      // query column within the panel

    const int qPerBlock      = 128;  // 8 waves * 16 queries
    const int blocksPerBatch = QTOT / qPerBlock;   // 256
    const int b = blockIdx.x / blocksPerBatch;
    const int q = (blockIdx.x % blocksPerBatch) * qPerBlock + wave * 16 + qn;

    const float* uvb   = uv   + (size_t)b * 2 * NPTS;
    const float* featb = feat + (size_t)b * CCH * NPTS;
    const int*   kp    = knn  + ((size_t)b * QTOT + q) * KNN;

    const float fx = (float)(q % IMW);
    const float fy = (float)(q / IMW);

    // ---- score-net hidden layer h_k[3] per neighbor -----------------------
    int   idx[KNN];
    float hk[KNN][3];
#pragma unroll
    for (int k = 0; k < KNN; ++k) {
        const int p = kp[k];
        idx[k] = p;
        const float ox = uvb[p]        - fx;
        const float oy = uvb[NPTS + p] - fy;
        const float nr = sqrtf(ox * ox + oy * oy);
#pragma unroll
        for (int j = 0; j < 3; ++j) {
            const float v = s_w1[j * 3 + 0] * ox + s_w1[j * 3 + 1] * oy
                          + s_w1[j * 3 + 2] * nr + s_b1[j];
            hk[k][j] = (v >= 0.f) ? v : 0.1f * v;
        }
    }

    // ---- sigmoid score * gathered feat -> final, packed as WMMA B operands
    v16h Bv0, Bv1;
#pragma unroll
    for (int cidx = 0; cidx < 2; ++cidx) {
#pragma unroll
        for (int j = 0; j < 16; ++j) {
            const int   c   = cidx * 32 + half * 16 + j;   // input channel
            const float w20 = s_w2[c * 3 + 0];
            const float w21 = s_w2[c * 3 + 1];
            const float w22 = s_w2[c * 3 + 2];
            const float bb  = s_b2[c];
            float acc = 0.f;
#pragma unroll
            for (int k = 0; k < KNN; ++k) {
                const float z  = w20 * hk[k][0] + w21 * hk[k][1] + w22 * hk[k][2] + bb;
                const float sc = 1.0f / (1.0f + __expf(-z));
                acc += sc * featb[c * NPTS + idx[k]];
            }
            if (cidx == 0) Bv0[j] = (_Float16)acc;
            else           Bv1[j] = (_Float16)acc;
        }
    }

    // ---- out_conv via v_wmma_f32_16x16x32_f16 -----------------------------
    v8f D[4];
#pragma unroll
    for (int t = 0; t < 4; ++t) D[t] = (v8f){0.f,0.f,0.f,0.f,0.f,0.f,0.f,0.f};

    const int kb = half * 16;  // contiguous K order: lanes 16-31 hold K=16..31 (matches B)
#pragma unroll
    for (int t = 0; t < 4; ++t) {
        v16h A0, A1;
#pragma unroll
        for (int e = 0; e < 16; ++e) {
            const int kk  = kb + e;                            // K within chunk
            const int row = t * 16 + qn;                       // A: M = lane%16
            A0[e] = s_wout[row * CCH + kk];                    // chunk0: in-chans 0..31
            A1[e] = s_wout[row * CCH + 32 + kk];               // chunk1: in-chans 32..63
        }
        D[t] = __builtin_amdgcn_wmma_f32_16x16x32_f16(false, A0, false, Bv0,
                                                      (short)0, D[t], false, false);
        D[t] = __builtin_amdgcn_wmma_f32_16x16x32_f16(false, A1, false, Bv1,
                                                      (short)0, D[t], false, false);
    }

    // ---- bias + leaky + store. D layout: N = lane%16, M = v + 8*half ------
    float* outb = out + (size_t)b * CCH * QTOT;
#pragma unroll
    for (int t = 0; t < 4; ++t) {
#pragma unroll
        for (int v = 0; v < 8; ++v) {
            const int ch = t * 16 + half * 8 + v;
            float val = D[t][v] + s_bout[ch];
            val = (val >= 0.f) ? val : 0.1f * val;
            outb[(size_t)ch * QTOT + q] = val;
        }
    }
}

// ---------------------------------------------------------------------------
extern "C" void kernel_launch(void* const* d_in, const int* in_sizes, int n_in,
                              void* d_out, int out_size, void* d_ws, size_t ws_size,
                              hipStream_t stream) {
    const float* uv    = (const float*)d_in[0];
    const float* feat  = (const float*)d_in[1];
    // d_in[2]=image_h, d_in[3]=image_w (baked-in constants 128/256)
    const float* w1    = (const float*)d_in[4];
    const float* b1    = (const float*)d_in[5];
    const float* w2    = (const float*)d_in[6];
    const float* b2    = (const float*)d_in[7];
    const float* w_out = (const float*)d_in[8];
    const float* b_out = (const float*)d_in[9];
    float* out = (float*)d_out;

    int* knn_idx = (int*)d_ws;   // needs BSZ*QTOT*KNN*4 = 768 KB scratch

    fai_knn_kernel<<<BSZ * (QTOT / 256), 256, 0, stream>>>(uv, knn_idx);
    fai_interp_kernel<<<BSZ * (QTOT / 128), 256, 0, stream>>>(
        uv, feat, knn_idx, w1, b1, w2, b2, w_out, b_out, out);
}
